// ARCTransformerEncoderLayerWithAttn_v1_33655363731496
// MI455X (gfx1250) — compile-verified, matrix operand order checked by reading, not run on hardware
//
#include <hip/hip_runtime.h>
#include <math.h>

typedef __attribute__((ext_vector_type(16))) _Float16 v16h;
typedef __attribute__((ext_vector_type(8)))  float    v8f;

#define B_    8
#define S_    1025
#define E_    1024
#define NH_   16
#define HD_   64
#define HID_  2730
#define FC1N_ 5460
#define K2P_  2752          // 2730 padded to multiple of 32 for fc2 K-loop
#define M_    8200          // B*S token rows
#define SCALE_ 0.125f       // 64^-0.5

union FragU { v16h v; uint4 u[2]; _Float16 h[16]; };

__device__ inline v8f wmma_f16(v16h a, v16h b, v8f c) {
  return __builtin_amdgcn_wmma_f32_16x16x32_f16(false, a, false, b, (short)0, c,
                                                false, false);
}

// ---- CDNA5 async global->LDS DMA (ASYNCcnt-tracked), via inline asm --------
__device__ inline void async_ld_b128(unsigned lds_off, const void* gaddr) {
  asm volatile("global_load_async_to_lds_b128 %0, %1, off"
               :
               : "v"(lds_off), "v"(gaddr)
               : "memory");
}
__device__ inline void wait_async0() {
  asm volatile("s_wait_asynccnt 0" ::: "memory");
}
// Low 32 bits of a generic pointer into LDS == byte offset in LDS aperture.
__device__ inline unsigned lds_off32(const void* p) {
  return (unsigned)(uintptr_t)p;
}

// Fragment load (A or B operand) from a [rows x 32] f16 tile with row stride 32.
// 16-bit operand lane layout: lanes 0-15 -> K 0..7 & 16..23, lanes 16-31 -> K 8..15 & 24..31.
__device__ inline v16h ld_frag_lds(const _Float16* p, int row, int lane) {
  FragU f;
  const _Float16* q = p + row * 32 + ((lane & 16) ? 8 : 0);
  f.u[0] = *(const uint4*)q;
  f.u[1] = *(const uint4*)(q + 16);
  return f.v;
}

// Fragment load from a global row pointer (contiguous 32 halves starting at rowp).
__device__ inline v16h ld_frag_row(const _Float16* rowp, int lane) {
  FragU f;
  const _Float16* q = rowp + ((lane & 16) ? 8 : 0);
  f.u[0] = *(const uint4*)q;
  f.u[1] = *(const uint4*)(q + 16);
  return f.v;
}

// ---------------- weight cast + transpose: fp32 [K,N] -> f16 [N,Kpad] -------
__global__ void cast_transpose_k(const float* __restrict__ W, _Float16* __restrict__ Wt,
                                 int K, int N, int Kpad) {
  int n = blockIdx.x;
  for (int k = threadIdx.x; k < Kpad; k += 256) {
    _Float16 v = (_Float16)0.f;
    if (k < K) v = (_Float16)W[(size_t)k * N + n];
    Wt[(size_t)n * Kpad + k] = v;
  }
}

// ---------------- RMSNorm (fp32 in) -> f16 out ------------------------------
__global__ __launch_bounds__(256) void rmsnorm_k(const float* __restrict__ x,
                                                 const float* __restrict__ g,
                                                 _Float16* __restrict__ out) {
  int row  = blockIdx.x;
  int tid  = threadIdx.x;
  int lane = tid & 31, wave = tid >> 5;
  __shared__ float red[8];
  const float* xr = x + (size_t)row * E_;
  float s = 0.f;
  for (int c = tid; c < E_; c += 256) { float v = xr[c]; s += v * v; }
  for (int off = 16; off >= 1; off >>= 1) s += __shfl_xor(s, off, 32);
  if (lane == 0) red[wave] = s;
  __syncthreads();
  float tot = 0.f;
  for (int w = 0; w < 8; ++w) tot += red[w];
  float rinv = rsqrtf(tot * (1.0f / E_) + 1e-6f);
  _Float16* orow = out + (size_t)row * E_;
  for (int c = tid; c < E_; c += 256) orow[c] = (_Float16)(xr[c] * rinv * g[c]);
}

// ---------------- WMMA GEMM: C[M,N] = A[M,Kpad] * Bt[N,Kpad]^T + bias -------
// Double-buffered LDS tiles fed by async global->LDS DMA.
// EPI 0: scatter qkv to f16 [3][B,NH,S,HD]
// EPI 1: outF = acc + bias + res   (fp32)
// EPI 2: outF = acc + bias         (fp32)
template <int EPI>
__global__ __launch_bounds__(256) void gemm_wmma_k(
    const _Float16* __restrict__ A, const _Float16* __restrict__ Bt,
    const float* __restrict__ bias, const float* __restrict__ res,
    float* __restrict__ outF,
    _Float16* __restrict__ outQ, _Float16* __restrict__ outK,
    _Float16* __restrict__ outV,
    int M, int N, int Kpad) {
  __shared__ _Float16 As[2][128 * 32];
  __shared__ _Float16 Bs[2][128 * 32];
  const int tid  = threadIdx.x;
  const int lane = tid & 31;
  const int wave = tid >> 5;
  const int wm   = wave >> 1;   // 0..3 -> 32-row slice
  const int wn   = wave & 1;    // 0..1 -> 64-col slice
  const int bm   = blockIdx.x * 128;
  const int bn   = blockIdx.y * 128;

  v8f zero = {};
  v8f acc[2][4];
#pragma unroll
  for (int i = 0; i < 2; ++i)
#pragma unroll
    for (int j = 0; j < 4; ++j) acc[i][j] = zero;

  // per-thread tile slice: 2 x B128 of A, 2 x B128 of B
  auto issue = [&](int buf, int k0) {
#pragma unroll
    for (int t = 0; t < 2; ++t) {
      int idx = tid + t * 256;          // 0..511
      int row = idx >> 2;
      int ch  = (idx & 3) * 8;
      int gm = bm + row; if (gm >= M) gm = M - 1;
      async_ld_b128(lds_off32(&As[buf][row * 32 + ch]),
                    A + (size_t)gm * Kpad + k0 + ch);
      int gn = bn + row; if (gn >= N) gn = N - 1;
      async_ld_b128(lds_off32(&Bs[buf][row * 32 + ch]),
                    Bt + (size_t)gn * Kpad + k0 + ch);
    }
  };

  const int nK = Kpad >> 5;
  issue(0, 0);
  for (int it = 0; it < nK; ++it) {
    const int cur = it & 1;
    wait_async0();            // only the current buffer's DMA is outstanding
    __syncthreads();          // publish LDS writes to all waves
    if (it + 1 < nK) issue(cur ^ 1, (it + 1) * 32);  // overlap next tile DMA

    v16h af[2], bf[4];
#pragma unroll
    for (int i = 0; i < 2; ++i)
      af[i] = ld_frag_lds(As[cur], wm * 32 + i * 16 + (lane & 15), lane);
#pragma unroll
    for (int j = 0; j < 4; ++j)
      bf[j] = ld_frag_lds(Bs[cur], wn * 64 + j * 16 + (lane & 15), lane);
#pragma unroll
    for (int i = 0; i < 2; ++i)
#pragma unroll
      for (int j = 0; j < 4; ++j) acc[i][j] = wmma_f16(af[i], bf[j], acc[i][j]);
    __syncthreads();          // all waves done reading before buffer reuse
  }

  const int colLane = lane & 15;
  const int rowOff  = (lane >> 4) * 8;
#pragma unroll
  for (int i = 0; i < 2; ++i)
#pragma unroll
    for (int j = 0; j < 4; ++j)
#pragma unroll
      for (int r = 0; r < 8; ++r) {
        int gm = bm + wm * 32 + i * 16 + rowOff + r;
        int gn = bn + wn * 64 + j * 16 + colLane;
        if (gm < M && gn < N) {
          float v = acc[i][j][r] + bias[gn];
          if constexpr (EPI == 1) {
            outF[(size_t)gm * N + gn] = v + res[(size_t)gm * N + gn];
          } else if constexpr (EPI == 2) {
            outF[(size_t)gm * N + gn] = v;
          } else {  // qkv scatter
            int kind = gn >> 10;
            int rem  = gn & 1023;
            int h    = rem >> 6;
            int d    = rem & 63;
            int b    = gm / S_;
            int s    = gm - b * S_;
            _Float16* dst = (kind == 0) ? outQ : ((kind == 1) ? outK : outV);
            dst[(((size_t)(b * NH_ + h)) * S_ + s) * HD_ + d] = (_Float16)v;
          }
        }
      }
}

// ---------------- RoPE in place on q/k f16 [B,NH,S,HD]; skip token 0 --------
__global__ __launch_bounds__(32) void rope_k(_Float16* __restrict__ Q,
                                             _Float16* __restrict__ K) {
  int idx = blockIdx.x;            // (b*NH+h)*S + s
  int s   = idx % S_;
  if (s < 1) return;               // NT = 1
  int lane = threadIdx.x;          // i = 0..31
  _Float16* base = ((blockIdx.y == 0) ? Q : K) + (size_t)idx * HD_;
  float pos  = (float)(s - 1);
  float freq = pos * __powf(10000.f, -(float)lane / 32.f);
  float c = __cosf(freq), sn = __sinf(freq);
  float t1 = (float)base[lane];
  float t2 = (float)base[lane + 32];
  base[lane]      = (_Float16)(t1 * c - t2 * sn);
  base[lane + 32] = (_Float16)(t2 * c + t1 * sn);
}

// ---------------- Flash attention: one wave per (b,h, 16-query tile) -------
// V tiles double-buffered in LDS via async DMA, overlapping QK^T + softmax.
__global__ __launch_bounds__(32) void attn_k(const _Float16* __restrict__ Q,
                                             const _Float16* __restrict__ Kt,
                                             const _Float16* __restrict__ V,
                                             _Float16* __restrict__ O) {
  const int bh   = blockIdx.x;          // 0..127
  const int b    = bh >> 4, h = bh & 15;
  const int q0   = blockIdx.y * 16;
  const int lane = threadIdx.x;
  const _Float16* qb = Q + (size_t)bh * S_ * HD_;
  const _Float16* kb = Kt + (size_t)bh * S_ * HD_;
  const _Float16* vb = V + (size_t)bh * S_ * HD_;

  int qrow = q0 + (lane & 15);
  if (qrow >= S_) qrow = S_ - 1;
  v16h qa0 = ld_frag_row(qb + (size_t)qrow * HD_, lane);       // d 0..31
  v16h qa1 = ld_frag_row(qb + (size_t)qrow * HD_ + 32, lane);  // d 32..63

  float mrun[8], lrun[8];
#pragma unroll
  for (int r = 0; r < 8; ++r) { mrun[r] = -3.0e38f; lrun[r] = 0.f; }
  v8f zero = {};
  v8f o[4];
#pragma unroll
  for (int j = 0; j < 4; ++j) o[j] = zero;

  __shared__ _Float16 P[16 * 32];
  __shared__ _Float16 Vs[2][32 * 64];   // double-buffered 32-key x 64-dim V tile
  const int prow = (lane >> 4) * 8;

  // per-lane: copy one 128B V row (key = kt + lane) as 8 async B128s
  auto issueV = [&](int buf, int kt2) {
    int key = kt2 + lane;
    if (key >= S_) key = S_ - 1;      // P is zero for OOB columns
    const _Float16* src = vb + (size_t)key * HD_;
#pragma unroll
    for (int c = 0; c < 8; ++c)
      async_ld_b128(lds_off32(&Vs[buf][lane * 64 + c * 8]), src + c * 8);
  };

  issueV(0, 0);
  const int nIt = (S_ + 31) / 32;     // 33
  for (int it = 0; it < nIt; ++it) {
    const int kt  = it * 32;
    const int cur = it & 1;
    // ---- scores: QK^T for 32 keys (V tile DMA still in flight) ----
    v8f sf[2];
#pragma unroll
    for (int g = 0; g < 2; ++g) {
      int n0  = kt + g * 16;
      int col = n0 + (lane & 15);
      int key = (col >= S_) ? (S_ - 1) : col;
      v16h kb0 = ld_frag_row(kb + (size_t)key * HD_, lane);
      v16h kb1 = ld_frag_row(kb + (size_t)key * HD_ + 32, lane);
      v8f t = wmma_f16(qa1, kb1, zero);
      t     = wmma_f16(qa0, kb0, t);
      bool oob = col >= S_;
#pragma unroll
      for (int r = 0; r < 8; ++r) t[r] = oob ? -1.0e30f : t[r] * SCALE_;
      sf[g] = t;
    }
    // ---- online softmax row stats (rows in VGPR index, cols in lanes) ----
    float corr[8], pv0[8], pv1[8];
#pragma unroll
    for (int r = 0; r < 8; ++r) {
      float v = fmaxf(sf[0][r], sf[1][r]);
      for (int off = 8; off >= 1; off >>= 1) v = fmaxf(v, __shfl_xor(v, off, 16));
      float mnew = fmaxf(mrun[r], v);
      float c    = __expf(mrun[r] - mnew);
      float p0   = __expf(sf[0][r] - mnew);
      float p1   = __expf(sf[1][r] - mnew);
      float srow = p0 + p1;
      for (int off = 8; off >= 1; off >>= 1) srow += __shfl_xor(srow, off, 16);
      lrun[r] = lrun[r] * c + srow;
      mrun[r] = mnew;
      corr[r] = c;
      pv0[r]  = p0;
      pv1[r]  = p1;
    }
    // ---- transpose P (C layout -> A layout) through LDS ----
#pragma unroll
    for (int r = 0; r < 8; ++r) {
      P[(prow + r) * 32 + (lane & 15)]      = (_Float16)pv0[r];
      P[(prow + r) * 32 + 16 + (lane & 15)] = (_Float16)pv1[r];
    }
    wait_async0();                          // V(cur) landed in LDS
    if (it + 1 < nIt) issueV(cur ^ 1, kt + 32);  // overlap next V tile DMA
    __syncthreads();                        // order P stores before reads
    v16h pf = ld_frag_lds(P, lane & 15, lane);
    // ---- PV accumulate from LDS V tile ----
    const int kbase = (lane & 16) ? 8 : 0;
#pragma unroll
    for (int j = 0; j < 4; ++j) {
      FragU vf;
      int d = j * 16 + (lane & 15);
#pragma unroll
      for (int jj = 0; jj < 8; ++jj) {
        vf.h[jj]     = Vs[cur][(kbase + jj) * 64 + d];
        vf.h[8 + jj] = Vs[cur][(kbase + 16 + jj) * 64 + d];
      }
#pragma unroll
      for (int r = 0; r < 8; ++r) o[j][r] *= corr[r];
      o[j] = wmma_f16(pf, vf.v, o[j]);
    }
    __syncthreads();   // drain LDS reads before P/V buffer reuse
  }

  // write normalized output to [b, s, h*64 + d] f16 (proj GEMM A layout)
#pragma unroll
  for (int j = 0; j < 4; ++j)
#pragma unroll
    for (int r = 0; r < 8; ++r) {
      int qr = q0 + prow + r;
      if (qr < S_) {
        float val = o[j][r] / lrun[r];
        O[((size_t)b * S_ + qr) * E_ + h * HD_ + j * 16 + (lane & 15)] =
            (_Float16)val;
      }
    }
}

// -------- depthwise 3x3 conv (skip token 0) + exact GELU * gate -> f16 -----
__global__ __launch_bounds__(256) void conv_gate_k(const float* __restrict__ h1,
                                                   const float* __restrict__ dww,
                                                   const float* __restrict__ dwb,
                                                   _Float16* __restrict__ gh) {
  int row = blockIdx.x;                         // 0..M_-1
  int c   = blockIdx.y * 256 + threadIdx.x;
  if (c >= K2P_) return;
  if (c >= HID_) { gh[(size_t)row * K2P_ + c] = (_Float16)0.f; return; }  // zero pad
  int b = row / S_;
  int s = row - b * S_;
  float xg;
  if (s == 0) {
    xg = h1[(size_t)row * FC1N_ + c];
  } else {
    int p = s - 1, rr = p >> 5, cc = p & 31;
    float acc = 0.f;
#pragma unroll
    for (int kr = 0; kr < 3; ++kr) {
      int r2 = rr + kr - 1;
      if (r2 < 0 || r2 >= 32) continue;
#pragma unroll
      for (int kc = 0; kc < 3; ++kc) {
        int c2 = cc + kc - 1;
        if (c2 < 0 || c2 >= 32) continue;
        acc += h1[((size_t)b * S_ + 1 + r2 * 32 + c2) * FC1N_ + c] *
               dww[(kr * 3 + kc) * HID_ + c];
      }
    }
    xg = acc + dwb[c];
  }
  float vg   = h1[(size_t)row * FC1N_ + HID_ + c];
  float gelu = 0.5f * xg * (1.0f + erff(xg * 0.70710678118654752f));
  gh[(size_t)row * K2P_ + c] = (_Float16)(gelu * vg);
}

// ---------------------------------------------------------------------------
extern "C" void kernel_launch(void* const* d_in, const int* in_sizes, int n_in,
                              void* d_out, int out_size, void* d_ws, size_t ws_size,
                              hipStream_t stream) {
  const float* x      = (const float*)d_in[0];
  const float* g1     = (const float*)d_in[1];
  const float* w_qkv  = (const float*)d_in[2];
  const float* b_qkv  = (const float*)d_in[3];
  const float* w_proj = (const float*)d_in[4];
  const float* b_proj = (const float*)d_in[5];
  const float* g2     = (const float*)d_in[6];
  const float* fc1_w  = (const float*)d_in[7];
  const float* fc1_b  = (const float*)d_in[8];
  const float* dw_w   = (const float*)d_in[9];
  const float* dw_b   = (const float*)d_in[10];
  const float* fc2_w  = (const float*)d_in[11];
  const float* fc2_b  = (const float*)d_in[12];
  float* out = (float*)d_out;

  char* base = (char*)d_ws;
  size_t off = 0;
  auto take = [&](size_t bytes) -> char* {
    char* p = base + off;
    off += (bytes + 255) & ~(size_t)255;
    return p;
  };
  _Float16* wqkv_t  = (_Float16*)take((size_t)3 * E_ * E_ * 2);
  _Float16* wproj_t = (_Float16*)take((size_t)E_ * E_ * 2);
  _Float16* wfc1_t  = (_Float16*)take((size_t)FC1N_ * E_ * 2);
  _Float16* wfc2_t  = (_Float16*)take((size_t)E_ * K2P_ * 2);
  _Float16* xn_h    = (_Float16*)take((size_t)M_ * E_ * 2);
  _Float16* q_h     = (_Float16*)take((size_t)B_ * NH_ * S_ * HD_ * 2);
  _Float16* k_h     = (_Float16*)take((size_t)B_ * NH_ * S_ * HD_ * 2);
  _Float16* v_h     = (_Float16*)take((size_t)B_ * NH_ * S_ * HD_ * 2);
  _Float16* att_h   = (_Float16*)take((size_t)M_ * E_ * 2);
  float*    x2      = (float*)take((size_t)M_ * E_ * 4);
  _Float16* xn2_h   = (_Float16*)take((size_t)M_ * E_ * 2);
  float*    h1      = (float*)take((size_t)M_ * FC1N_ * 4);
  _Float16* g_h     = (_Float16*)take((size_t)M_ * K2P_ * 2);

  // 1) weight casts (fp32 [K,N] -> f16 [N,Kpad] transposed)
  cast_transpose_k<<<3 * E_, 256, 0, stream>>>(w_qkv, wqkv_t, E_, 3 * E_, E_);
  cast_transpose_k<<<E_, 256, 0, stream>>>(w_proj, wproj_t, E_, E_, E_);
  cast_transpose_k<<<FC1N_, 256, 0, stream>>>(fc1_w, wfc1_t, E_, FC1N_, E_);
  cast_transpose_k<<<E_, 256, 0, stream>>>(fc2_w, wfc2_t, HID_, E_, K2P_);

  // 2) rmsnorm(x, g1) -> f16
  rmsnorm_k<<<M_, 256, 0, stream>>>(x, g1, xn_h);

  // 3) qkv GEMM + bias, scatter to q/k/v f16 [B,NH,S,HD]
  gemm_wmma_k<0><<<dim3(65, 24), 256, 0, stream>>>(
      xn_h, wqkv_t, b_qkv, nullptr, nullptr, q_h, k_h, v_h, M_, 3 * E_, E_);

  // 4) RoPE in place on q and k (skip first token)
  rope_k<<<dim3(B_ * NH_ * S_, 2), 32, 0, stream>>>(q_h, k_h);

  // 5) attention -> att_h f16 [M, E]
  attn_k<<<dim3(B_ * NH_, (S_ + 15) / 16), 32, 0, stream>>>(q_h, k_h, v_h, att_h);

  // 6) proj GEMM + bias + residual(x) -> x2 fp32
  gemm_wmma_k<1><<<dim3(65, 8), 256, 0, stream>>>(
      att_h, wproj_t, b_proj, x, x2, nullptr, nullptr, nullptr, M_, E_, E_);

  // 7) rmsnorm(x2, g2) -> f16
  rmsnorm_k<<<M_, 256, 0, stream>>>(x2, g2, xn2_h);

  // 8) fc1 GEMM + bias -> h1 fp32 [M, 5460]
  gemm_wmma_k<2><<<dim3(65, 43), 256, 0, stream>>>(
      xn2_h, wfc1_t, fc1_b, nullptr, h1, nullptr, nullptr, nullptr, M_, FC1N_, E_);

  // 9) depthwise conv + GELU * gate -> g_h f16 [M, 2752] (zero padded)
  conv_gate_k<<<dim3(M_, (K2P_ + 255) / 256), 256, 0, stream>>>(h1, dw_w, dw_b, g_h);

  // 10) fc2 GEMM + bias + residual(x2) -> out fp32
  gemm_wmma_k<1><<<dim3(65, 8), 256, 0, stream>>>(
      g_h, wfc2_t, fc2_b, x2, out, nullptr, nullptr, nullptr, M_, E_, K2P_);
}
